// DualAttention_3642132267704
// MI455X (gfx1250) — matrix operand order repaired and checked by reading, not compiled, not run on hardware
//
#include <hip/hip_runtime.h>
#include <hip/hip_bf16.h>

typedef __attribute__((ext_vector_type(16))) __bf16 v16bf;
typedef __attribute__((ext_vector_type(8)))  float  v8f;

#define S_DIM 2048
#define B_DIM 16
#define H2_DIM 2048
#define V_DIM 1024
#define M_TOT (S_DIM * B_DIM)      // 32768
#define K_TOT (2 * H2_DIM)         // 4096
#define M_TILE 32                  // rows per block (2 WMMA row-tiles)
#define K_CHUNK 512
#define LDS_K (K_CHUNK + 8)        // +8 bf16 pad -> 4-bank shift per row

__device__ __forceinline__ unsigned short f2bf(float f) {
    unsigned u = __float_as_uint(f);
    unsigned r = u + 0x7FFFu + ((u >> 16) & 1u);   // round-to-nearest-even
    return (unsigned short)(r >> 16);
}

// ---------------------------------------------------------------------------
// Kernel 1: build WT[v][k] bf16 (k = [Ww rows ; Wz rows]) + combined bias
// ---------------------------------------------------------------------------
__global__ void prep_kernel(const float* __restrict__ Ww, const float* __restrict__ Wz,
                            const float* __restrict__ bw, const float* __restrict__ bz,
                            unsigned short* __restrict__ WT, float* __restrict__ bias) {
    int idx = blockIdx.x * 256 + threadIdx.x;      // 1024*4096 total
    int n = idx >> 12;
    int k = idx & (K_TOT - 1);
    const float* src = (k < H2_DIM) ? Ww : Wz;
    int kk = k & (H2_DIM - 1);
    WT[idx] = f2bf(src[(size_t)kk * V_DIM + n]);
    if (idx < V_DIM) bias[idx] = bw[idx] + bz[idx];
}

// ---------------------------------------------------------------------------
// Kernel 2: fused dual-GEMM (bf16 WMMA, f32 accum) + bias + tanh + dot(Vw)
// One block = 32 rows of M, full N=1024. 16 waves; wave owns 64 columns and
// 2 row-tiles -> 8 WMMAs per 8 B-loads (B reused across row tiles).
// ---------------------------------------------------------------------------
__global__ __launch_bounds__(512) void gemm_tanh_dot_kernel(
    const float* __restrict__ hidden, const float* __restrict__ z,
    const unsigned short* __restrict__ WT, const float* __restrict__ bias,
    const float* __restrict__ Vw, const float* __restrict__ vb,
    const float* __restrict__ w_a, float* __restrict__ u)
{
    __shared__ unsigned short A_lds[M_TILE][LDS_K];
    __shared__ float u_lds[M_TILE];

    const int tid  = threadIdx.x;
    const int wave = tid >> 5;
    const int lane = tid & 31;
    const int ln   = lane & 15;       // column / row-within-half
    const int lhi  = lane >> 4;       // lane half selector
    const int m0   = blockIdx.x * M_TILE;
    const int nbase = wave * 64;

    if (tid < M_TILE) u_lds[tid] = 0.0f;

    v8f c00 = {}, c01 = {}, c02 = {}, c03 = {};
    v8f c10 = {}, c11 = {}, c12 = {}, c13 = {};

    union AV { v16bf v; int4 q[2]; };

    for (int kc = 0; kc < K_TOT; kc += K_CHUNK) {
        __syncthreads();
        // ---- stage A chunk: 32 rows x 512 k, f32 -> bf16 into LDS ----
        const float* src = (kc < H2_DIM) ? hidden : z;
        const int koff = kc & (H2_DIM - 1);
        #pragma unroll
        for (int it = 0; it < 8; ++it) {
            int i  = tid + it * 512;           // 4096 float4s total
            int r  = i >> 7;                   // 128 float4 per row-chunk
            int k4 = i & 127;
            float4 f = *(const float4*)(src + (size_t)(m0 + r) * H2_DIM + koff + k4 * 4);
            unsigned p0 = (unsigned)f2bf(f.x) | ((unsigned)f2bf(f.y) << 16);
            unsigned p1 = (unsigned)f2bf(f.z) | ((unsigned)f2bf(f.w) << 16);
            *(uint2*)&A_lds[r][k4 * 4] = make_uint2(p0, p1);
        }
        __syncthreads();

        // ---- K loop: 32 per WMMA ----
        for (int kb = 0; kb < K_CHUNK; kb += 32) {
            // A fragments (16 consecutive K per lane, half-lanes split K by 16)
            AV a0, a1;
            const unsigned short* ap0 = &A_lds[ln][kb + lhi * 16];
            const unsigned short* ap1 = &A_lds[16 + ln][kb + lhi * 16];
            a0.q[0] = *(const int4*)ap0;  a0.q[1] = *(const int4*)(ap0 + 8);
            a1.q[0] = *(const int4*)ap1;  a1.q[1] = *(const int4*)(ap1 + 8);

            // B fragments (lane=col, 16 consecutive K per lane, same k order as A)
            const int kgB = kc + kb + lhi * 16;
            AV b0, b1, b2, b3;
            const unsigned short* bp0 = WT + ((size_t)(nbase + ln)      << 12) + kgB;
            const unsigned short* bp1 = WT + ((size_t)(nbase + 16 + ln) << 12) + kgB;
            const unsigned short* bp2 = WT + ((size_t)(nbase + 32 + ln) << 12) + kgB;
            const unsigned short* bp3 = WT + ((size_t)(nbase + 48 + ln) << 12) + kgB;
            b0.q[0] = *(const int4*)bp0;  b0.q[1] = *(const int4*)(bp0 + 8);
            b1.q[0] = *(const int4*)bp1;  b1.q[1] = *(const int4*)(bp1 + 8);
            b2.q[0] = *(const int4*)bp2;  b2.q[1] = *(const int4*)(bp2 + 8);
            b3.q[0] = *(const int4*)bp3;  b3.q[1] = *(const int4*)(bp3 + 8);

            c00 = __builtin_amdgcn_wmma_f32_16x16x32_bf16(false, a0.v, false, b0.v, (short)0, c00, false, false);
            c01 = __builtin_amdgcn_wmma_f32_16x16x32_bf16(false, a0.v, false, b1.v, (short)0, c01, false, false);
            c02 = __builtin_amdgcn_wmma_f32_16x16x32_bf16(false, a0.v, false, b2.v, (short)0, c02, false, false);
            c03 = __builtin_amdgcn_wmma_f32_16x16x32_bf16(false, a0.v, false, b3.v, (short)0, c03, false, false);
            c10 = __builtin_amdgcn_wmma_f32_16x16x32_bf16(false, a1.v, false, b0.v, (short)0, c10, false, false);
            c11 = __builtin_amdgcn_wmma_f32_16x16x32_bf16(false, a1.v, false, b1.v, (short)0, c11, false, false);
            c12 = __builtin_amdgcn_wmma_f32_16x16x32_bf16(false, a1.v, false, b2.v, (short)0, c12, false, false);
            c13 = __builtin_amdgcn_wmma_f32_16x16x32_bf16(false, a1.v, false, b3.v, (short)0, c13, false, false);
        }
    }

    // ---- epilogue: bias + w_a*alpha, tanh, * Vw[n], reduce over n ----
    const float wa2 = w_a[0] * 0.5f;
    float p[16];
    #pragma unroll
    for (int j = 0; j < 16; ++j) p[j] = 0.0f;

    auto epi = [&](const v8f& c, int rt, int t) {
        int n = nbase + t * 16 + ln;
        float bb = bias[n] + wa2;
        float vw = Vw[n];
        #pragma unroll
        for (int j = 0; j < 8; ++j)
            p[rt * 8 + j] += tanhf(c[j] + bb) * vw;
    };
    epi(c00, 0, 0); epi(c01, 0, 1); epi(c02, 0, 2); epi(c03, 0, 3);
    epi(c10, 1, 0); epi(c11, 1, 1); epi(c12, 1, 2); epi(c13, 1, 3);

    // reduce across the 16 lanes of each half (same m rows)
    #pragma unroll
    for (int rt = 0; rt < 2; ++rt) {
        #pragma unroll
        for (int j = 0; j < 8; ++j) {
            float v = p[rt * 8 + j];
            v += __shfl_xor(v, 1);
            v += __shfl_xor(v, 2);
            v += __shfl_xor(v, 4);
            v += __shfl_xor(v, 8);
            if (ln == 0) atomicAdd(&u_lds[rt * 16 + lhi * 8 + j], v);  // m = rt*16 + 8*lhi + j
        }
    }
    __syncthreads();
    if (tid < M_TILE) u[m0 + tid] = u_lds[tid] + vb[0];
}

// ---------------------------------------------------------------------------
// Kernel 3: softmax over sequence axis (dim 0). u is [S, B] with m = s*B+b.
// ---------------------------------------------------------------------------
__global__ void softmax_kernel(const float* __restrict__ u, float* __restrict__ out) {
    const int b = blockIdx.x;       // 16 blocks
    const int tid = threadIdx.x;    // 256 threads
    __shared__ float red[256];

    float mx = -3.4e38f;
    for (int s = tid; s < S_DIM; s += 256) mx = fmaxf(mx, u[s * B_DIM + b]);
    red[tid] = mx; __syncthreads();
    for (int off = 128; off > 0; off >>= 1) {
        if (tid < off) red[tid] = fmaxf(red[tid], red[tid + off]);
        __syncthreads();
    }
    mx = red[0]; __syncthreads();

    float sm = 0.0f;
    for (int s = tid; s < S_DIM; s += 256) sm += expf(u[s * B_DIM + b] - mx);
    red[tid] = sm; __syncthreads();
    for (int off = 128; off > 0; off >>= 1) {
        if (tid < off) red[tid] += red[tid + off];
        __syncthreads();
    }
    const float inv = 1.0f / red[0];

    for (int s = tid; s < S_DIM; s += 256)
        out[s * B_DIM + b] = expf(u[s * B_DIM + b] - mx) * inv;
}

// ---------------------------------------------------------------------------
extern "C" void kernel_launch(void* const* d_in, const int* in_sizes, int n_in,
                              void* d_out, int out_size, void* d_ws, size_t ws_size,
                              hipStream_t stream) {
    const float* hidden = (const float*)d_in[0];
    const float* z      = (const float*)d_in[1];
    const float* Ww     = (const float*)d_in[2];
    const float* bw     = (const float*)d_in[3];
    const float* Wz     = (const float*)d_in[4];
    const float* bz     = (const float*)d_in[5];
    const float* Vw     = (const float*)d_in[6];
    const float* vb     = (const float*)d_in[7];
    const float* w_a    = (const float*)d_in[8];
    float* out = (float*)d_out;

    char* ws = (char*)d_ws;
    unsigned short* WT = (unsigned short*)ws;                        // 1024*4096*2 = 8 MB
    float* bias        = (float*)(ws + (size_t)V_DIM * K_TOT * 2);   // 4 KB
    float* u           = (float*)(ws + (size_t)V_DIM * K_TOT * 2 + V_DIM * 4); // 128 KB

    prep_kernel<<<(V_DIM * K_TOT) / 256, 256, 0, stream>>>(Ww, Wz, bw, bz, WT, bias);
    gemm_tanh_dot_kernel<<<M_TOT / M_TILE, 512, 0, stream>>>(hidden, z, WT, bias, Vw, vb, w_a, u);
    softmax_kernel<<<B_DIM, 256, 0, stream>>>(u, out);
}
